// Decoder_81595788689581
// MI455X (gfx1250) — matrix operand order repaired and checked by reading, not compiled, not run on hardware
//
#include <hip/hip_runtime.h>

typedef __attribute__((ext_vector_type(16))) __bf16 v16bf;
typedef __attribute__((ext_vector_type(8)))  __bf16 v8bf;
typedef __attribute__((ext_vector_type(8)))  float  v8f;

static __device__ __forceinline__ __bf16 f2bf(float f) {
  // round-to-nearest-even f32 -> bf16
  unsigned int u = __float_as_uint(f);
  unsigned int r = (u + 0x7FFFu + ((u >> 16) & 1u)) >> 16;
  unsigned short s = (unsigned short)r;
  __bf16 b;
  __builtin_memcpy(&b, &s, sizeof(b));
  return b;
}

// ---------------------------------------------------------------------------
// C[M,N] = A[M,K](bf16,row-major) * B[N,K](bf16,row-major)^T + bias[N]
// Each wave computes a 16x64 output tile via 4x v_wmma_f32_16x16x32_bf16
// per 32-wide K chunk. blockDim.x = 256 (8 waves). Requires:
//   M % 16 == 0, N % 64 == 0, K % 32 == 0, gridDim.x*8 == (M/16)*(N/64)
// Writes f32 C (if non-null) and/or bf16 Cbf (if non-null).
// ---------------------------------------------------------------------------
__global__ void gemm_bf16_wmma(const __bf16* __restrict__ A,
                               const __bf16* __restrict__ Bw,
                               const float*  __restrict__ bias,
                               float*        __restrict__ C,
                               __bf16*       __restrict__ Cbf,
                               int M, int N, int K) {
  const int lane = threadIdx.x & 31;
  const int wave = blockIdx.x * (blockDim.x >> 5) + (threadIdx.x >> 5);
  const int tilesN = N >> 6;                 // 64-wide N tiles
  const int tm = wave / tilesN;
  const int tn = wave - tm * tilesN;
  if (tm * 16 >= M) return;                  // wave-uniform

  // A fragment addressing (16-bit A 16x32, contiguous k order chosen to match B, not the ISA table order):
  // lane L: row = L%16; K elems {g..g+15}, g = (L/16)*16 (matches B)
  const int rowA  = tm * 16 + (lane & 15);
  const int kGrpA = (lane >> 4) << 4;        // 0 or 16
  // B fragment addressing (16-bit B 32x16): lane L: col = L%16,
  // K elems (L/16)*16 + [0..15]  -> one contiguous 32B load per subtile
  const int nSub  = lane & 15;
  const int kGrpB = (lane >> 4) << 4;        // 0 or 16
  const int colBase = tn << 6;

  const __bf16* Arow = A + (size_t)rowA * K + kGrpA;

  v8f acc0 = {}, acc1 = {}, acc2 = {}, acc3 = {};
  for (int k0 = 0; k0 < K; k0 += 32) {
    v8bf alo = *reinterpret_cast<const v8bf*>(Arow + k0);
    v8bf ahi = *reinterpret_cast<const v8bf*>(Arow + k0 + 8);
    v16bf a;
#pragma unroll
    for (int i = 0; i < 8; ++i) { a[i] = alo[i]; a[i + 8] = ahi[i]; }

    const __bf16* Bbase = Bw + (size_t)(colBase + nSub) * K + k0 + kGrpB;
    v16bf b0 = *reinterpret_cast<const v16bf*>(Bbase);
    v16bf b1 = *reinterpret_cast<const v16bf*>(Bbase + (size_t)16 * K);
    v16bf b2 = *reinterpret_cast<const v16bf*>(Bbase + (size_t)32 * K);
    v16bf b3 = *reinterpret_cast<const v16bf*>(Bbase + (size_t)48 * K);

    acc0 = __builtin_amdgcn_wmma_f32_16x16x32_bf16(false, a, false, b0, (short)0, acc0, false, false);
    acc1 = __builtin_amdgcn_wmma_f32_16x16x32_bf16(false, a, false, b1, (short)0, acc1, false, false);
    acc2 = __builtin_amdgcn_wmma_f32_16x16x32_bf16(false, a, false, b2, (short)0, acc2, false, false);
    acc3 = __builtin_amdgcn_wmma_f32_16x16x32_bf16(false, a, false, b3, (short)0, acc3, false, false);
  }

  // C/D layout: VGPR v -> row = tm*16 + v + 8*(lane>=16), col = base + lane%16
  const int mBase = tm * 16 + ((lane >> 4) << 3);
#pragma unroll
  for (int j = 0; j < 4; ++j) {
    v8f accv = (j == 0) ? acc0 : (j == 1) ? acc1 : (j == 2) ? acc2 : acc3;
    const int n = colBase + j * 16 + nSub;
    const float bv = bias ? bias[n] : 0.0f;
#pragma unroll
    for (int v = 0; v < 8; ++v) {
      const float val = accv[v] + bv;
      const size_t idx = (size_t)(mBase + v) * N + n;
      if (C)   C[idx] = val;
      if (Cbf) Cbf[idx] = f2bf(val);
    }
  }
}

// ---------------------------------------------------------------------------
// Per-step attention: e[b,t'] = W3 . tanh(w1[b]+w2[b,t']); a = softmax(e);
// c[b,:] = sum_t' a[t'] * dx[b,t',:]; pack X = [c | x_t | 0pad] as bf16[128,160]
// One block per batch row, 256 threads (8 waves).
// ---------------------------------------------------------------------------
__global__ void attention_step(const float* __restrict__ w1,   // [128,512]
                               const float* __restrict__ w2,   // [128,48,512]
                               const float* __restrict__ W3,   // [512]
                               const float* __restrict__ dx,   // [128,48,128]
                               const float* __restrict__ x,    // [128,48,3]
                               __bf16* __restrict__ Xbf,       // [128,160]
                               float* __restrict__ a_out,      // [128,48] or null
                               int t) {
  __shared__ float sE[48];
  __shared__ float sA[48];
  __shared__ float sMS[2];
  const int b = blockIdx.x;
  const int tid = threadIdx.x;
  const int lane = tid & 31;
  const int w = tid >> 5;

  const float* w1row = w1 + (size_t)b * 512;
  for (int tt = w; tt < 48; tt += 8) {
    const float* w2row = w2 + (size_t)(b * 48 + tt) * 512;
    float p = 0.0f;
    for (int h = lane; h < 512; h += 32)
      p += tanhf(w1row[h] + w2row[h]) * W3[h];
    for (int off = 16; off > 0; off >>= 1)
      p += __shfl_xor(p, off, 32);
    if (lane == 0) sE[tt] = p;
  }
  __syncthreads();
  if (tid == 0) {
    float m = sE[0];
    for (int i = 1; i < 48; ++i) m = fmaxf(m, sE[i]);
    float s = 0.0f;
    for (int i = 0; i < 48; ++i) s += expf(sE[i] - m);
    sMS[0] = m; sMS[1] = 1.0f / s;
  }
  __syncthreads();
  if (tid < 48) {
    const float av = expf(sE[tid] - sMS[0]) * sMS[1];
    sA[tid] = av;
    if (a_out) a_out[b * 48 + tid] = av;
  }
  __syncthreads();
  if (tid < 128) {
    const float* dxb = dx + (size_t)b * 48 * 128 + tid;
    float c = 0.0f;
#pragma unroll 4
    for (int tt = 0; tt < 48; ++tt) c += sA[tt] * dxb[(size_t)tt * 128];
    Xbf[b * 160 + tid] = f2bf(c);
  } else if (tid < 131) {
    Xbf[b * 160 + tid] = f2bf(x[(size_t)b * 144 + t * 3 + (tid - 128)]);
  } else if (tid < 160) {
    Xbf[b * 160 + tid] = f2bf(0.0f);
  }
}

// GRU gate nonlinearity (PyTorch order r,z,n). 128*512 elements.
__global__ void gru_gates(const float* __restrict__ gi, const float* __restrict__ gh,
                          float* __restrict__ h, __bf16* __restrict__ hbf) {
  const int idx = blockIdx.x * blockDim.x + threadIdx.x;  // 0..65535
  const int b = idx >> 9;
  const int hh = idx & 511;
  const size_t base = (size_t)b * 1536;
  const float ir  = gi[base + hh],        hr = gh[base + hh];
  const float iz  = gi[base + 512 + hh],  hz = gh[base + 512 + hh];
  const float in_ = gi[base + 1024 + hh], hn = gh[base + 1024 + hh];
  const float r = 1.0f / (1.0f + expf(-(ir + hr)));
  const float z = 1.0f / (1.0f + expf(-(iz + hz)));
  const float n = tanhf(in_ + r * hn);
  const float hnew = (1.0f - z) * n + z * h[idx];
  h[idx] = hnew;
  hbf[idx] = f2bf(hnew);
}

__global__ void cast_f32_bf16(const float* __restrict__ src, __bf16* __restrict__ dst, int n) {
  const int i = blockIdx.x * blockDim.x + threadIdx.x;
  if (i < n) dst[i] = f2bf(src[i]);
}

// W4 [512,131] f32 -> [512,160] bf16, zero-padded along K
__global__ void pad_W4(const float* __restrict__ W4, __bf16* __restrict__ W4bf) {
  const int i = blockIdx.x * blockDim.x + threadIdx.x;  // 512*160
  const int nrow = i / 160, kk = i - nrow * 160;
  W4bf[i] = (kk < 131) ? f2bf(W4[nrow * 131 + kk]) : f2bf(0.0f);
}

__global__ void init_h(const float* __restrict__ h0, float* __restrict__ h,
                       __bf16* __restrict__ hbf, int n) {
  const int i = blockIdx.x * blockDim.x + threadIdx.x;
  if (i < n) { h[i] = h0[i]; hbf[i] = f2bf(h0[i]); }
}

// re[b] = h[b,:] . Wfc + bfc
__global__ void final_fc(const float* __restrict__ h, const float* __restrict__ Wfc,
                         const float* __restrict__ bfc, float* __restrict__ out) {
  const int b = threadIdx.x;  // 128 threads
  const float* hr = h + (size_t)b * 512;
  float s = 0.0f;
  for (int k = 0; k < 512; ++k) s += hr[k] * Wfc[k];
  out[b] = s + bfc[0];
}

extern "C" void kernel_launch(void* const* d_in, const int* in_sizes, int n_in,
                              void* d_out, int out_size, void* d_ws, size_t ws_size,
                              hipStream_t stream) {
  (void)in_sizes; (void)n_in; (void)out_size; (void)ws_size;
  const float* dx  = (const float*)d_in[0];   // [128,48,128]
  const float* x   = (const float*)d_in[1];   // [128,48,3]
  const float* h0  = (const float*)d_in[2];   // [128,512]
  const float* W1  = (const float*)d_in[3];   // [512,512]
  const float* b1  = (const float*)d_in[4];
  const float* W2  = (const float*)d_in[5];   // [512,128]
  const float* b2  = (const float*)d_in[6];
  const float* W3  = (const float*)d_in[7];   // [1,512]
  // d_in[8] = b3: softmax-invariant, unused
  const float* W4  = (const float*)d_in[9];   // [512,131]
  const float* b4  = (const float*)d_in[10];
  const float* Wih = (const float*)d_in[11];  // [1536,512]
  const float* Whh = (const float*)d_in[12];  // [1536,512]
  const float* bih = (const float*)d_in[13];
  const float* bhh = (const float*)d_in[14];
  const float* Wfc = (const float*)d_in[15];  // [1,512]
  const float* bfc = (const float*)d_in[16];
  float* out = (float*)d_out;  // [0..127]=re, [128..6271]=a_f (b-major, t-minor)

  char* ws = (char*)d_ws;
  size_t off = 0;
  auto carve = [&](size_t bytes) -> void* {
    off = (off + 255) & ~(size_t)255;
    void* p = ws + off;
    off += bytes;
    return p;
  };
  __bf16* W1bf  = (__bf16*)carve((size_t)512 * 512 * 2);
  __bf16* W2bf  = (__bf16*)carve((size_t)512 * 128 * 2);
  __bf16* W4bf  = (__bf16*)carve((size_t)512 * 160 * 2);
  __bf16* Wihbf = (__bf16*)carve((size_t)1536 * 512 * 2);
  __bf16* Whhbf = (__bf16*)carve((size_t)1536 * 512 * 2);
  __bf16* dxbf  = (__bf16*)carve((size_t)6144 * 128 * 2);
  float*  w2f   = (float*)carve((size_t)6144 * 512 * 4);
  float*  w1f   = (float*)carve((size_t)128 * 512 * 4);
  __bf16* Xbf   = (__bf16*)carve((size_t)128 * 160 * 2);
  __bf16* y0bf  = (__bf16*)carve((size_t)128 * 512 * 2);
  float*  gif   = (float*)carve((size_t)128 * 1536 * 4);
  float*  ghf   = (float*)carve((size_t)128 * 1536 * 4);
  float*  hf    = (float*)carve((size_t)128 * 512 * 4);
  __bf16* hbf   = (__bf16*)carve((size_t)128 * 512 * 2);

  // ---- prologue: weight/input conversion + step-invariant w2 GEMM ----
  cast_f32_bf16<<<262144 / 256, 256, 0, stream>>>(W1, W1bf, 262144);
  cast_f32_bf16<<<65536 / 256, 256, 0, stream>>>(W2, W2bf, 65536);
  pad_W4<<<81920 / 256, 256, 0, stream>>>(W4, W4bf);
  cast_f32_bf16<<<786432 / 256, 256, 0, stream>>>(Wih, Wihbf, 786432);
  cast_f32_bf16<<<786432 / 256, 256, 0, stream>>>(Whh, Whhbf, 786432);
  cast_f32_bf16<<<786432 / 256, 256, 0, stream>>>(dx, dxbf, 786432);
  init_h<<<65536 / 256, 256, 0, stream>>>(h0, hf, hbf, 65536);
  // w2[b,t,:] = dx[b,t,:] @ W2^T + b2  (M=6144, N=512, K=128) -> 3072 waves
  gemm_bf16_wmma<<<384, 256, 0, stream>>>(dxbf, W2bf, b2, w2f, nullptr, 6144, 512, 128);

  // ---- sequential scan over T=48 steps ----
  for (int t = 0; t < 48; ++t) {
    // w1 = h @ W1^T + b1   (128x512x512) -> 64 waves
    gemm_bf16_wmma<<<8, 256, 0, stream>>>(hbf, W1bf, b1, w1f, nullptr, 128, 512, 512);
    attention_step<<<128, 256, 0, stream>>>(w1f, w2f, W3, dx, x, Xbf,
                                            (t == 47) ? (out + 128) : nullptr, t);
    // y0 = [c|x_t|pad] @ W4pad^T + b4  (128x512x160), bf16 out only
    gemm_bf16_wmma<<<8, 256, 0, stream>>>(Xbf, W4bf, b4, nullptr, y0bf, 128, 512, 160);
    // gi = y0 @ W_ih^T + b_ih ; gh = h @ W_hh^T + b_hh  (128x1536x512 each)
    gemm_bf16_wmma<<<24, 256, 0, stream>>>(y0bf, Wihbf, bih, gif, nullptr, 128, 1536, 512);
    gemm_bf16_wmma<<<24, 256, 0, stream>>>(hbf, Whhbf, bhh, ghf, nullptr, 128, 1536, 512);
    gru_gates<<<256, 256, 0, stream>>>(gif, ghf, hf, hbf);
  }

  // ---- epilogue: re = h_f @ Wfc^T + bfc ----
  final_fc<<<1, 128, 0, stream>>>(hf, Wfc, bfc, out);
}
